// AFGSA_2113123909815
// MI455X (gfx1250) — compile-verified, matrix operand order checked by reading, not run on hardware
//
#include <hip/hip_runtime.h>

// ---------------------------------------------------------------------------
// AFGSA windowed halo-attention for MI455X (gfx1250, wave32, WMMA bf16)
// Round 2: + async-to-LDS (ASYNCcnt) staging of GEMM weight tiles (double-
// buffered) and attention q tiles; vectorized k/v global reads.
// ---------------------------------------------------------------------------

typedef __bf16 bf16;
typedef __attribute__((ext_vector_type(16))) __bf16 v16bf;
typedef __attribute__((ext_vector_type(8)))  __bf16 v8bf;
typedef __attribute__((ext_vector_type(8)))  float  v8f;

#define CH     256
#define IMG    128
#define PIX    16384              // 128*128
#define NPIXT  65536              // 4*128*128
#define HEADS  4
#define DHEAD  64
#define WINSZ  14
#define NPOS   196                // 14*14
#define KPAD   224                // next multiple of 32 >= 196

// ---- CDNA5 async global->LDS copy (16B per lane), tracked by ASYNCcnt.
__device__ inline void async_b128(const void* gsrc, unsigned lds_byte_off) {
  asm volatile("global_load_async_to_lds_b128 %0, %1, off"
               :: "v"(lds_byte_off), "v"(gsrc) : "memory");
}
__device__ inline void wait_async_0() {
#if __has_builtin(__builtin_amdgcn_s_wait_asynccnt)
  __builtin_amdgcn_s_wait_asynccnt(0);
#else
  asm volatile("s_wait_asynccnt 0x0" ::: "memory");
#endif
}
__device__ inline void wait_async_le2() {
#if __has_builtin(__builtin_amdgcn_s_wait_asynccnt)
  __builtin_amdgcn_s_wait_asynccnt(2);
#else
  asm volatile("s_wait_asynccnt 0x2" ::: "memory");
#endif
}

// ---- load a 16x32 bf16 fragment (A: rows x K, or B: cols x K) from
// row-major storage with `stride` bf16 elements between rows.
// ISA 16-bit A/B layout: lane<16 holds row(lane), K {0..7,16..23};
// lane>=16 holds row(lane-16), K {8..15,24..31}. Two b128 loads per lane.
__device__ inline v16bf load_frag_row(const bf16* base, int stride) {
  const int lane = threadIdx.x & 31;
  const int r    = lane & 15;
  const int kb   = (lane >> 4) << 3;          // 0 or 8
  const bf16* p  = base + (size_t)r * stride + kb;
  v8bf lo = *(const v8bf*)(p);
  v8bf hi = *(const v8bf*)(p + 16);
  v16bf o;
#pragma unroll
  for (int i = 0; i < 8; ++i) { o[i] = lo[i]; o[i + 8] = hi[i]; }
  return o;
}

__device__ inline v8f wmma_bf16(v16bf a, v16bf b, v8f c) {
  return __builtin_amdgcn_wmma_f32_16x16x32_bf16(
      /*neg_a=*/false, a, /*neg_b=*/false, b,
      /*c_mod=*/(short)0, c, /*reuse_a=*/false, /*reuse_b=*/false);
}

// ---------------------------------------------------------------------------
// fp32 -> bf16 elementwise convert (weights)
// ---------------------------------------------------------------------------
__global__ void cvt_f32_bf16(const float* __restrict__ src,
                             bf16* __restrict__ dst, int n) {
  int i = blockIdx.x * blockDim.x + threadIdx.x;
  if (i < n) dst[i] = (bf16)src[i];
}

// ---------------------------------------------------------------------------
// NCHW f32 -> pixel-major bf16. dstCat gets channel block [catOff, catOff+256),
// dstN (optional) gets a 256-wide copy (noisy for the V projection).
// ---------------------------------------------------------------------------
__global__ void pack_nchw(const float* __restrict__ src,
                          bf16* __restrict__ dstCat,
                          bf16* __restrict__ dstN, int catOff) {
  size_t i = (size_t)blockIdx.x * blockDim.x + threadIdx.x; // over 4*256*16384
  int p = (int)(i & 16383);
  int c = (int)((i >> 14) & 255);
  int b = (int)(i >> 22);
  bf16 v = (bf16)src[i];
  size_t pix = ((size_t)b << 14) | (size_t)p;
  dstCat[pix * 512 + catOff + c] = v;
  if (dstN) dstN[pix * 256 + c] = v;
}

// ---------------------------------------------------------------------------
// GEMM: out[M][256] = f( in[M][K] * W[256][K]^T + bias ) in bf16.
// 128 threads = 4 waves; block tile 64 rows x 64 cols; wave tile 16 x 64.
// Weight tile (64 cols x 32 K = 4KB) staged in LDS via async-to-LDS, double
// buffered: DMA of step i+1 overlaps WMMA of step i.
// ---------------------------------------------------------------------------
__global__ __launch_bounds__(128)
void gemm_bf16(const bf16* __restrict__ in, const bf16* __restrict__ W,
               const float* __restrict__ bias, bf16* __restrict__ out,
               int K, int doRelu, float scale) {
  __shared__ __align__(16) bf16 Bbuf[2][64 * 32];

  const int tid   = threadIdx.x;
  const int lane  = tid & 31;
  const int wave  = tid >> 5;
  const int rbase = blockIdx.x * 64 + wave * 16;
  const int cbase = blockIdx.y * 64;

  // async fill of one 64x32 weight tile: 256 16B chunks, 2 per thread
  auto fill = [&](int bufidx, int kc) {
#pragma unroll
    for (int u = 0; u < 2; ++u) {
      int c   = tid + u * 128;
      int col = c >> 2;
      int ko  = (c & 3) << 3;
      async_b128(W + (size_t)(cbase + col) * K + kc + ko,
                 (unsigned)(size_t)(&Bbuf[bufidx][col * 32 + ko]));
    }
  };

  v8f acc[4] = {};
  const int nk = K >> 5;
  fill(0, 0);
  for (int i = 0; i < nk; ++i) {
    if (i + 1 < nk) {
      fill((i + 1) & 1, (i + 1) << 5);  // prefetch next tile (2 ops/thread)
      wait_async_le2();                 // wait for current tile only
    } else {
      wait_async_0();
    }
    __syncthreads();
    const bf16* Bt = Bbuf[i & 1];
    v16bf a = load_frag_row(in + (size_t)rbase * K + (i << 5), K);
#pragma unroll
    for (int t = 0; t < 4; ++t) {
      v16bf b = load_frag_row(Bt + t * 16 * 32, 32);
      acc[t] = wmma_bf16(a, b, acc[t]);
    }
    __syncthreads();                    // tile fully consumed before overwrite
  }

  // C layout: VGPR g, lanes 0-15 -> row g, lanes 16-31 -> row g+8; col = lane&15
  const int rrow = rbase + (lane >> 4) * 8;
#pragma unroll
  for (int t = 0; t < 4; ++t) {
    int col = cbase + t * 16 + (lane & 15);
    float bv = bias ? bias[col] : 0.0f;
#pragma unroll
    for (int g = 0; g < 8; ++g) {
      float v = acc[t][g] + bv;
      if (doRelu) v = fmaxf(v, 0.0f);
      v *= scale;
      out[(size_t)(rrow + g) * 256 + col] = (bf16)v;
    }
  }
}

// ---------------------------------------------------------------------------
// Windowed halo attention. 1 block = (window, head); 128 threads = 4 waves.
// LDS: qs 64x64 | ks 224x64 (rel-biased, zero-padded) | vsT 64x224 | aw 4x16x224
// ---------------------------------------------------------------------------
__global__ __launch_bounds__(128)
void attn_win(const bf16* __restrict__ Q, const bf16* __restrict__ Km,
              const bf16* __restrict__ V, const float* __restrict__ relh,
              const float* __restrict__ relw, float* __restrict__ out) {
  extern __shared__ __align__(16) char smem[];
  bf16* qs  = (bf16*)(smem);                        // 64*64*2   = 8192 B
  bf16* ks  = (bf16*)(smem + 8192);                 // 224*64*2  = 28672 B
  bf16* vsT = (bf16*)(smem + 8192 + 28672);         // 64*224*2  = 28672 B
  bf16* aw  = (bf16*)(smem + 8192 + 2 * 28672);     // 4*16*224*2= 28672 B

  const int tid  = threadIdx.x;
  const int lane = tid & 31;
  const int wave = tid >> 5;

  const int head = blockIdx.x & 3;
  const int win  = blockIdx.x >> 2;
  const int b    = win >> 8;
  const int wy   = (win >> 4) & 15;
  const int wx   = win & 15;

  // ---- q tile: 64 query pixels x 64 head channels, async global->LDS
  {
    int qi = tid >> 1;
    int d0 = (tid & 1) << 5;
    int yy = wy * 8 + (qi >> 3), xx = wx * 8 + (qi & 7);
    const bf16* src = Q + (((size_t)b * IMG + yy) * IMG + xx) * CH + head * DHEAD + d0;
    bf16* dst = qs + qi * DHEAD + d0;
#pragma unroll
    for (int u = 0; u < 4; ++u)
      async_b128(src + u * 8, (unsigned)(size_t)(dst + u * 8));
  }

  // ---- k (+rel bias, zero-padded halo) and v (transposed into vsT)
  for (int p = tid; p < KPAD; p += 128) {
    int r = p / WINSZ, cc = p % WINSZ;              // junk for p>=196 (masked)
    bool valid = (p < NPOS);
    int y = wy * 8 - 3 + r, x = wx * 8 - 3 + cc;
    bool img = valid && (y >= 0) && (y < IMG) && (x >= 0) && (x < IMG);
    size_t base = (((size_t)b * IMG + y) * IMG + x) * CH + head * DHEAD;
    const bf16* kp = Km + base;
    const bf16* vp = V + base;
    float kf[DHEAD], vf[DHEAD];
    if (img) {
#pragma unroll
      for (int u = 0; u < 8; ++u) {
        v8bf k8 = *(const v8bf*)(kp + u * 8);   // b128 loads
        v8bf v8 = *(const v8bf*)(vp + u * 8);
#pragma unroll
        for (int j = 0; j < 8; ++j) {
          kf[u * 8 + j] = (float)k8[j];
          vf[u * 8 + j] = (float)v8[j];
        }
      }
    } else {
#pragma unroll
      for (int dd = 0; dd < DHEAD; ++dd) { kf[dd] = 0.0f; vf[dd] = 0.0f; }
    }
#pragma unroll
    for (int dd = 0; dd < DHEAD; ++dd) {
      float kv = kf[dd];
      if (valid)
        kv += (dd < 32) ? relh[r * 32 + dd] : relw[cc * 32 + (dd - 32)];
      ks[p * DHEAD + dd] = (bf16)kv;
      vsT[dd * KPAD + p] = (bf16)vf[dd];
    }
  }
  wait_async_0();                                   // q tile landed in LDS
  __syncthreads();

  // ---- sim = q . k^T : wave owns 16 query rows, 14 col tiles of 16 keys
  const int qb = wave * 16;
  v8f acc[14] = {};
#pragma unroll
  for (int kc = 0; kc < DHEAD; kc += 32) {
    v16bf a = load_frag_row(qs + qb * DHEAD + kc, DHEAD);
#pragma unroll
    for (int t = 0; t < 14; ++t) {
      v16bf bb = load_frag_row(ks + (t * 16) * DHEAD + kc, DHEAD);
      acc[t] = wmma_bf16(a, bb, acc[t]);
    }
  }

  // ---- mask padded key columns (>=196)
  {
    int cl = lane & 15;
#pragma unroll
    for (int g = 0; g < 8; ++g) {
      if (cl >= 4) acc[12][g] = -1e30f;             // cols 196..207
      acc[13][g] = -1e30f;                          // cols 208..223
    }
  }

  // ---- softmax per query row (row spread over 16 lanes in each half-wave)
#pragma unroll
  for (int g = 0; g < 8; ++g) {
    float m = -1e30f;
#pragma unroll
    for (int t = 0; t < 14; ++t) m = fmaxf(m, acc[t][g]);
    m = fmaxf(m, __shfl_xor(m, 1, 32));
    m = fmaxf(m, __shfl_xor(m, 2, 32));
    m = fmaxf(m, __shfl_xor(m, 4, 32));
    m = fmaxf(m, __shfl_xor(m, 8, 32));
    float s = 0.0f;
#pragma unroll
    for (int t = 0; t < 14; ++t) {
      float e = __expf(acc[t][g] - m);
      acc[t][g] = e;
      s += e;
    }
    s += __shfl_xor(s, 1, 32);
    s += __shfl_xor(s, 2, 32);
    s += __shfl_xor(s, 4, 32);
    s += __shfl_xor(s, 8, 32);
    float inv = 1.0f / s;
#pragma unroll
    for (int t = 0; t < 14; ++t) acc[t][g] *= inv;
  }

  // ---- spill attn (C layout) to LDS as bf16 for A-fragment re-layout
  {
    int rl = qb + (lane >> 4) * 8;
    int cl = lane & 15;
#pragma unroll
    for (int t = 0; t < 14; ++t)
#pragma unroll
      for (int g = 0; g < 8; ++g)
        aw[(size_t)(rl + g) * KPAD + t * 16 + cl] = (bf16)acc[t][g];
  }
  // aw region is per-wave and LDS ops are in-order within a wave.

  // ---- out = attn . v : 4 col tiles (d), K = 224 in 7 chunks
  v8f o[4] = {};
#pragma unroll
  for (int kc = 0; kc < KPAD; kc += 32) {
    v16bf a = load_frag_row(aw + (size_t)qb * KPAD + kc, KPAD);
#pragma unroll
    for (int t = 0; t < 4; ++t) {
      v16bf bb = load_frag_row(vsT + (t * 16) * KPAD + kc, KPAD);
      o[t] = wmma_bf16(a, bb, o[t]);
    }
  }

  // ---- scatter to NCHW f32 output: channel = head*64+dd
  {
    int cl = lane & 15;
    int rh = (lane >> 4) * 8;
#pragma unroll
    for (int t = 0; t < 4; ++t) {
      int dd = t * 16 + cl;
      size_t cb = (((size_t)b * CH) + head * DHEAD + dd) * PIX;
#pragma unroll
      for (int g = 0; g < 8; ++g) {
        int qi = qb + rh + g;
        int yy = wy * 8 + (qi >> 3), xx = wx * 8 + (qi & 7);
        out[cb + (size_t)yy * IMG + xx] = o[t][g];
      }
    }
  }
}

// ---------------------------------------------------------------------------
extern "C" void kernel_launch(void* const* d_in, const int* in_sizes, int n_in,
                              void* d_out, int out_size, void* d_ws, size_t ws_size,
                              hipStream_t stream) {
  const float* noisy = (const float*)d_in[0];
  const float* aux   = (const float*)d_in[1];
  const float* w_map = (const float*)d_in[2];
  const float* b_map = (const float*)d_in[3];
  const float* w_q   = (const float*)d_in[4];
  const float* w_k   = (const float*)d_in[5];
  const float* w_v   = (const float*)d_in[6];
  const float* rel_h = (const float*)d_in[7];
  const float* rel_w = (const float*)d_in[8];
  float* out = (float*)d_out;

  char* ws = (char*)d_ws;
  size_t off = 0;
  auto alloc = [&](size_t bytes) -> char* {
    char* p = ws + off;
    off += (bytes + 255) & ~(size_t)255;
    return p;
  };
  bf16* Xcat = (bf16*)alloc((size_t)NPIXT * 512 * 2);  // concat(noisy,aux) bf16
  bf16* Xn   = (bf16*)alloc((size_t)NPIXT * 256 * 2);  // noisy bf16 (for V)
  bf16* Naux = (bf16*)alloc((size_t)NPIXT * 256 * 2);
  bf16* Qb   = (bf16*)alloc((size_t)NPIXT * 256 * 2);
  bf16* Kb   = (bf16*)alloc((size_t)NPIXT * 256 * 2);
  bf16* Vb   = (bf16*)alloc((size_t)NPIXT * 256 * 2);
  bf16* WmapB = (bf16*)alloc((size_t)256 * 512 * 2);
  bf16* WqB   = (bf16*)alloc((size_t)256 * 256 * 2);
  bf16* WkB   = (bf16*)alloc((size_t)256 * 256 * 2);
  bf16* WvB   = (bf16*)alloc((size_t)256 * 256 * 2);

  // weights -> bf16
  cvt_f32_bf16<<<(131072 + 255) / 256, 256, 0, stream>>>(w_map, WmapB, 131072);
  cvt_f32_bf16<<<(65536 + 255) / 256, 256, 0, stream>>>(w_q, WqB, 65536);
  cvt_f32_bf16<<<(65536 + 255) / 256, 256, 0, stream>>>(w_k, WkB, 65536);
  cvt_f32_bf16<<<(65536 + 255) / 256, 256, 0, stream>>>(w_v, WvB, 65536);

  // NCHW -> pixel-major bf16
  pack_nchw<<<65536, 256, 0, stream>>>(noisy, Xcat, Xn, 0);
  pack_nchw<<<65536, 256, 0, stream>>>(aux, Xcat, nullptr, 256);

  dim3 gg(NPIXT / 64, 256 / 64);
  // n_aux = relu(Xcat @ Wmap^T + b)
  gemm_bf16<<<gg, 128, 0, stream>>>(Xcat, WmapB, b_map, Naux, 512, 1, 1.0f);
  // q (pre-scaled by d^-0.5 = 0.125), k, v
  gemm_bf16<<<gg, 128, 0, stream>>>(Naux, WqB, nullptr, Qb, 256, 0, 0.125f);
  gemm_bf16<<<gg, 128, 0, stream>>>(Naux, WkB, nullptr, Kb, 256, 0, 1.0f);
  gemm_bf16<<<gg, 128, 0, stream>>>(Xn,   WvB, nullptr, Vb, 256, 0, 1.0f);

  // windowed attention: 4 batches * 256 windows * 4 heads = 4096 blocks
  size_t shmem = 8192 + 3 * 28672;  // 94208 B
  attn_win<<<4096, 128, shmem, stream>>>(Qb, Kb, Vb, rel_h, rel_w, out);
}
